// Block_62526133895402
// MI455X (gfx1250) — compile-verified, matrix operand order checked by reading, not run on hardware
//
#include <hip/hip_runtime.h>
#include <math.h>

// ---- problem constants ----
#define Bc    64
#define G1c   128
#define Sc    32
#define GSc   16
#define Dc    384
#define Tc    10
#define Hc    6
#define HDc   64
#define N1c   139            // G1 + 1 + T
#define M1c   (Bc * N1c)     // 8896 rows, stage 1
#define Gc    138            // G1 + T
#define BSc   (Bc * Sc)      // 2048 groups
#define M2c   (BSc * GSc)    // 32768 rows, stage 2
#define SCALEc 0.125f        // HD^-0.5

typedef __attribute__((ext_vector_type(16))) _Float16 v16h;
typedef __attribute__((ext_vector_type(8)))  _Float16 v8h;
typedef __attribute__((ext_vector_type(8)))  float    v8f;

__device__ __forceinline__ float gelu_f(float x) {
  return 0.5f * x * (1.0f + erff(x * 0.70710678118654752f));
}

// =====================================================================
// GEMM: C[M,N] = epilogue( A[M,K] @ W[N,K]^T ),  A/W pre-converted f16.
// V_WMMA_F32_16X16X32_F16, f32 accumulate. One wave -> 16x(NT*16) C
// strip; NT is compile-time so the K-loop is branch-free (no spills,
// accumulators pinned). NT=4 for N%64==0, NT=1 for N=16 adapters.
// Requirements: K % 32 == 0, M % 16 == 0, N % (NT*16) == 0,
// lda/ldw multiples of 8 halves (16B-aligned b128 loads).
// epilogue: v = alpha * act(gemm + bias) + s1*R1 + s2*R2  (R*/C stride ldc)
// Writes f32 C (if non-null) and/or f16 C16 (if non-null).
// =====================================================================
template <int NT>
__global__ void gemm_wmma_f16(const _Float16* __restrict__ A, int lda,
                              const _Float16* __restrict__ W, int ldw,
                              const float* __restrict__ bias,
                              float* __restrict__ C,
                              _Float16* __restrict__ C16, int ldc,
                              const float* __restrict__ R1, float s1,
                              const float* __restrict__ R2, float s2,
                              int M, int K, float alpha, int act) {
  const int lane = threadIdx.x & 31;
  const int half = lane >> 4;       // 0: lanes 0-15, 1: lanes 16-31
  const int lm   = lane & 15;
  const int tn0  = blockIdx.x * (NT * 16);                  // base column
  const int tm   = blockIdx.y * blockDim.y + threadIdx.y;   // M tile
  if (tm * 16 >= M) return;           // uniform per wave -> EXEC all-ones

  // 16-bit A 16x32 fragment (ISA 7.12.2): per lane, two contiguous
  // 8-half K chunks at half*8 and 16+half*8 (each one b128 load).
  const _Float16* Ap = A + (size_t)(tm * 16 + lm) * lda + half * 8;
  const _Float16* Wp[NT];
#pragma unroll
  for (int nt = 0; nt < NT; ++nt)
    Wp[nt] = W + (size_t)(tn0 + nt * 16 + lm) * ldw + half * 8;

  v8f acc[NT];
#pragma unroll
  for (int nt = 0; nt < NT; ++nt) acc[nt] = {};

  union Frag { v16h v; v8h h[2]; };

  for (int k0 = 0; k0 < K; k0 += 32) {
    Frag a;
    a.h[0] = *reinterpret_cast<const v8h*>(Ap);
    a.h[1] = *reinterpret_cast<const v8h*>(Ap + 16);
    Ap += 32;
    __builtin_prefetch(Ap + 32, 0, 0);   // speculative global_prefetch
#pragma unroll
    for (int nt = 0; nt < NT; ++nt) {
      Frag b;
      b.h[0] = *reinterpret_cast<const v8h*>(Wp[nt]);
      b.h[1] = *reinterpret_cast<const v8h*>(Wp[nt] + 16);
      Wp[nt] += 32;
      acc[nt] = __builtin_amdgcn_wmma_f32_16x16x32_f16(
          false, a.v, false, b.v, (short)0, acc[nt], false, false);
    }
  }

  // C/D 16x16 f32 layout: VGPR r -> M = r + 8*half, N = lane&15
#pragma unroll
  for (int nt = 0; nt < NT; ++nt) {
    const int n = tn0 + nt * 16 + lm;
#pragma unroll
    for (int r = 0; r < 8; ++r) {
      const int m = tm * 16 + r + half * 8;
      float v = acc[nt][r];
      if (bias) v += bias[n];
      if (act)  v = gelu_f(v);
      v *= alpha;
      const size_t o = (size_t)m * ldc + n;
      if (R1) v += s1 * R1[o];
      if (R2) v += s2 * R2[o];
      if (C)   C[o] = v;
      if (C16) C16[o] = (_Float16)v;
    }
  }
}

// =====================================================================
// Utility kernels
// =====================================================================
__global__ void zero_kernel(float* __restrict__ p, int n) {
  const int i = blockIdx.x * blockDim.x + threadIdx.x;
  if (i < n) p[i] = 0.0f;
}

__global__ void f32to16_kernel(const float* __restrict__ in,
                               _Float16* __restrict__ out, int n) {
  const int i = blockIdx.x * blockDim.x + threadIdx.x;
  if (i < n) out[i] = (_Float16)in[i];
}

// Pad (N,16) f32 weights to (N,32) f16 with zero columns 16..31
__global__ void padw_kernel(const float* __restrict__ w,
                            _Float16* __restrict__ wp, int nrows) {
  const int i = blockIdx.x * blockDim.x + threadIdx.x;
  if (i >= nrows * 32) return;
  const int k = i & 31;
  const int r = i >> 5;
  wp[i] = (k < 16) ? (_Float16)w[r * 16 + k] : (_Float16)0.0f;
}

// =====================================================================
// Build x_cat[B,139,D] = concat(x[:, :1], prompt, x[:, 1:])
// =====================================================================
__global__ void build_xcat_kernel(const float* __restrict__ x,
                                  const float* __restrict__ prompt,
                                  float* __restrict__ xcat) {
  const int i = blockIdx.x * blockDim.x + threadIdx.x;
  if (i >= M1c * Dc) return;
  const int d = i % Dc;
  const int r = i / Dc;
  const int n = r % N1c;
  const int b = r / N1c;
  float v;
  if (n == 0)        v = x[((size_t)b * (G1c + 1)) * Dc + d];
  else if (n <= Tc)  v = prompt[(size_t)(n - 1) * Dc + d];
  else               v = x[((size_t)b * (G1c + 1) + (n - Tc)) * Dc + d];
  xcat[i] = v;
}

// =====================================================================
// Row LayerNorm over D=384; one block (128 threads) per row; f16 out
// (consumed only by GEMM A operand).
// =====================================================================
__global__ void layernorm_kernel(const float* __restrict__ in,
                                 const float* __restrict__ g,
                                 const float* __restrict__ b,
                                 _Float16* __restrict__ out, int rows) {
  const int r = blockIdx.x;
  if (r >= rows) return;
  const float* xr = in + (size_t)r * Dc;
  __shared__ float s1[128], s2[128];
  float a = 0.0f, q = 0.0f;
  for (int d = threadIdx.x; d < Dc; d += 128) {
    const float v = xr[d];
    a += v; q += v * v;
  }
  s1[threadIdx.x] = a; s2[threadIdx.x] = q;
  __syncthreads();
  for (int s = 64; s > 0; s >>= 1) {
    if (threadIdx.x < s) { s1[threadIdx.x] += s1[threadIdx.x + s];
                           s2[threadIdx.x] += s2[threadIdx.x + s]; }
    __syncthreads();
  }
  const float mean = s1[0] * (1.0f / Dc);
  const float var  = s2[0] * (1.0f / Dc) - mean * mean;
  const float inv  = rsqrtf(var + 1e-5f);
  for (int d = threadIdx.x; d < Dc; d += 128)
    out[(size_t)r * Dc + d] = (_Float16)((xr[d] - mean) * inv * g[d] + b[d]);
}

// =====================================================================
// Stage-1 attention: one block per (b,h,n); probs f32 -> d_out,
// attention output f16 (consumed only by proj GEMM).
// qkv layout: [B*N1, 3*D] rows; q at h*HD, k at D+h*HD, v at 2D+h*HD.
// =====================================================================
__global__ void attn1_kernel(const float* __restrict__ qkv,
                             float* __restrict__ probs,
                             _Float16* __restrict__ attnout) {
  const int bid = blockIdx.x;
  const int n  = bid % N1c;
  const int bh = bid / N1c;
  const int h  = bh % Hc;
  const int b  = bh / Hc;
  __shared__ float sc[N1c];
  __shared__ float q[HDc];
  __shared__ float red[128];
  const float* qr = qkv + ((size_t)(b * N1c + n)) * (3 * Dc) + h * HDc;
  for (int d = threadIdx.x; d < HDc; d += 128) q[d] = qr[d];
  __syncthreads();
  for (int m = threadIdx.x; m < N1c; m += 128) {
    const float* kr = qkv + ((size_t)(b * N1c + m)) * (3 * Dc) + Dc + h * HDc;
    float s = 0.0f;
    for (int d = 0; d < HDc; ++d) s += q[d] * kr[d];
    sc[m] = s * SCALEc;
  }
  __syncthreads();
  float mx = -1e30f;
  for (int m = threadIdx.x; m < N1c; m += 128) mx = fmaxf(mx, sc[m]);
  red[threadIdx.x] = mx; __syncthreads();
  for (int s = 64; s > 0; s >>= 1) {
    if (threadIdx.x < s) red[threadIdx.x] = fmaxf(red[threadIdx.x], red[threadIdx.x + s]);
    __syncthreads();
  }
  mx = red[0]; __syncthreads();
  float sum = 0.0f;
  for (int m = threadIdx.x; m < N1c; m += 128) {
    const float e = __expf(sc[m] - mx);
    sc[m] = e; sum += e;
  }
  red[threadIdx.x] = sum; __syncthreads();
  for (int s = 64; s > 0; s >>= 1) {
    if (threadIdx.x < s) red[threadIdx.x] += red[threadIdx.x + s];
    __syncthreads();
  }
  const float inv = 1.0f / red[0];
  __syncthreads();
  float* pr = probs + ((size_t)((b * Hc + h) * N1c + n)) * N1c;
  for (int m = threadIdx.x; m < N1c; m += 128) {
    sc[m] *= inv;
    pr[m] = sc[m];
  }
  __syncthreads();
  for (int d = threadIdx.x; d < HDc; d += 128) {
    float o = 0.0f;
    for (int m = 0; m < N1c; ++m)
      o += sc[m] * qkv[((size_t)(b * N1c + m)) * (3 * Dc) + 2 * Dc + h * HDc + d];
    attnout[((size_t)(b * N1c + n)) * Dc + h * HDc + d] = (_Float16)o;
  }
}

// =====================================================================
// Stage-2 attention: seq len 16. One block (256 thr) per (group, head).
// Output f16 (consumed only by proj1 GEMM).
// =====================================================================
__global__ void attn2_kernel(const float* __restrict__ qkv,
                             _Float16* __restrict__ attnout) {
  const int h = blockIdx.x % Hc;
  const int g = blockIdx.x / Hc;   // group in [0, B*S)
  const int tn  = threadIdx.x >> 4;   // n in 0..15
  const int tmd = threadIdx.x & 15;   // m (scores) / d-chunk (output)
  __shared__ float P[16][16];
  const size_t base = (size_t)g * GSc * (3 * Dc);
  const float* qr = qkv + base + (size_t)tn  * (3 * Dc) + h * HDc;
  const float* kr = qkv + base + (size_t)tmd * (3 * Dc) + Dc + h * HDc;
  float s = 0.0f;
  for (int d = 0; d < HDc; ++d) s += qr[d] * kr[d];
  P[tn][tmd] = s * SCALEc;
  __syncthreads();
  if (tmd == 0) {
    float mx = -1e30f;
    for (int m = 0; m < 16; ++m) mx = fmaxf(mx, P[tn][m]);
    float sum = 0.0f;
    for (int m = 0; m < 16; ++m) { const float e = __expf(P[tn][m] - mx); P[tn][m] = e; sum += e; }
    const float inv = 1.0f / sum;
    for (int m = 0; m < 16; ++m) P[tn][m] *= inv;
  }
  __syncthreads();
#pragma unroll
  for (int dd = 0; dd < 4; ++dd) {
    const int d = tmd * 4 + dd;
    float o = 0.0f;
    for (int m = 0; m < 16; ++m)
      o += P[tn][m] * qkv[base + (size_t)m * (3 * Dc) + 2 * Dc + h * HDc + d];
    attnout[((size_t)(g * GSc + tn)) * Dc + h * HDc + d] = (_Float16)o;
  }
}

// =====================================================================
// Gather rows of prompt_x (== x2 with row 0 per batch dropped).
// prompt_x row v  ->  x2 row (v/138)*139 + (v%138) + 1
// =====================================================================
__global__ void gather_kernel(const float* __restrict__ x2,
                              const int* __restrict__ idx,
                              float* __restrict__ dst, int rows) {
  const int i = blockIdx.x * blockDim.x + threadIdx.x;
  if (i >= rows * Dc) return;
  const int d = i % Dc;
  const int r = i / Dc;
  const int v = idx[r];
  const int src = (v / Gc) * N1c + (v % Gc) + 1;
  dst[i] = x2[(size_t)src * Dc + d];
}

// =====================================================================
// lc = GELU(BN(max over group)) ; vis_x = lc + 0.3 * x_centers
// =====================================================================
__global__ void maxbn_kernel(const float* __restrict__ xn2,
                             const float* __restrict__ bng,
                             const float* __restrict__ bnb,
                             const float* __restrict__ bnm,
                             const float* __restrict__ bnv,
                             const float* __restrict__ xcent,
                             float* __restrict__ visx) {
  const int i = blockIdx.x * blockDim.x + threadIdx.x;
  if (i >= BSc * Dc) return;
  const int d = i % Dc;
  const int g = i / Dc;
  float mx = -1e30f;
  for (int j = 0; j < GSc; ++j)
    mx = fmaxf(mx, xn2[((size_t)g * GSc + j) * Dc + d]);
  float lc = (mx - bnm[d]) * rsqrtf(bnv[d] + 1e-5f) * bng[d] + bnb[d];
  lc = gelu_f(lc);
  visx[i] = lc + 0.3f * xcent[i];
}

// =====================================================================
// propagate: de_neighbors == S, so the argsort is a no-op — weighted
// mean over all 32 neighbors. One block (128 thr) per (b, point).
// =====================================================================
__global__ void propagate_kernel(const float* __restrict__ c1,
                                 const float* __restrict__ c2,
                                 const float* __restrict__ x2,   // xg rows live here
                                 const float* __restrict__ visx,
                                 float* __restrict__ out) {
  const int bi = blockIdx.x;
  const int i = bi % G1c;
  const int b = bi / G1c;
  __shared__ float w[Sc];
  __shared__ float winv;
  if (threadIdx.x < Sc) {
    const float dx = c1[(b * G1c + i) * 3 + 0] - c2[(b * Sc + threadIdx.x) * 3 + 0];
    const float dy = c1[(b * G1c + i) * 3 + 1] - c2[(b * Sc + threadIdx.x) * 3 + 1];
    const float dz = c1[(b * G1c + i) * 3 + 2] - c2[(b * Sc + threadIdx.x) * 3 + 2];
    w[threadIdx.x] = 1.0f / (dx * dx + dy * dy + dz * dz + 1e-8f);
  }
  __syncthreads();
  if (threadIdx.x == 0) {
    float s = 0.0f;
    for (int j = 0; j < Sc; ++j) s += w[j];
    winv = 1.0f / s;
  }
  __syncthreads();
  const float inv = winv;
  const float* xr = x2 + ((size_t)(b * N1c + Tc + 1 + i)) * Dc;  // xg row
  for (int d = threadIdx.x; d < Dc; d += 128) {
    float acc = 0.0f;
    for (int j = 0; j < Sc; ++j)
      acc += w[j] * visx[((size_t)(b * Sc + j)) * Dc + d];
    out[((size_t)(b * G1c + i)) * Dc + d] = xr[d] + 0.3f * acc * inv;
  }
}

// =====================================================================
// x3[B,139,D]: rows 0..T from x2 (cls + prompt), rows T+1.. from prop.
// Dual write: f32 (residual) + f16 (GEMM A operand).
// =====================================================================
__global__ void build_x3_kernel(const float* __restrict__ x2,
                                const float* __restrict__ prop,
                                float* __restrict__ x3,
                                _Float16* __restrict__ x316) {
  const int i = blockIdx.x * blockDim.x + threadIdx.x;
  if (i >= M1c * Dc) return;
  const int d = i % Dc;
  const int r = i / Dc;
  const int n = r % N1c;
  const int b = r / N1c;
  const float v = (n <= Tc) ? x2[((size_t)(b * N1c + n)) * Dc + d]
                            : prop[((size_t)(b * G1c + (n - Tc - 1))) * Dc + d];
  x3[i] = v;
  x316[i] = (_Float16)v;
}

// =====================================================================
// Final: out[B,129,D] = concat(y[:, :1], y[:, T+1:])
// =====================================================================
__global__ void final_copy_kernel(const float* __restrict__ y,
                                  float* __restrict__ out) {
  const int i = blockIdx.x * blockDim.x + threadIdx.x;
  if (i >= Bc * (G1c + 1) * Dc) return;
  const int d = i % Dc;
  const int r = i / Dc;
  const int g = r % (G1c + 1);
  const int b = r / (G1c + 1);
  const int srcn = (g == 0) ? 0 : (Tc + g);
  out[i] = y[((size_t)(b * N1c + srcn)) * Dc + d];
}

// =====================================================================
// Host launch
// =====================================================================
static inline void launch_gemm(const _Float16* A, int lda, const _Float16* W, int ldw,
                               const float* bias, float* C, _Float16* C16, int ldc,
                               const float* R1, float s1,
                               const float* R2, float s2,
                               int M, int N, int K, float alpha, int act,
                               hipStream_t stream) {
  dim3 block(32, 4);
  if (N % 64 == 0) {
    dim3 grid(N / 64, (M + 63) / 64);
    gemm_wmma_f16<4><<<grid, block, 0, stream>>>(A, lda, W, ldw, bias, C, C16, ldc,
                                                 R1, s1, R2, s2, M, K, alpha, act);
  } else {
    dim3 grid(N / 16, (M + 63) / 64);
    gemm_wmma_f16<1><<<grid, block, 0, stream>>>(A, lda, W, ldw, bias, C, C16, ldc,
                                                 R1, s1, R2, s2, M, K, alpha, act);
  }
}

extern "C" void kernel_launch(void* const* d_in, const int* in_sizes, int n_in,
                              void* d_out, int out_size, void* d_ws, size_t ws_size,
                              hipStream_t stream) {
  const float* x          = (const float*)d_in[0];
  // d_in[1] mask        : unused by reference mha
  const float* center1    = (const float*)d_in[2];
  const float* center2    = (const float*)d_in[3];
  // d_in[4] neighborhood: result unused in reference
  const int*   idx        = (const int*)d_in[5];
  const int*   center_idx = (const int*)d_in[6];
  // d_in[7] group_size, d_in[8] layer_id: unused
  const float* ln1_g = (const float*)d_in[9];
  const float* ln1_b = (const float*)d_in[10];
  const float* ln2_g = (const float*)d_in[11];
  const float* ln2_b = (const float*)d_in[12];
  const float* qkv_w = (const float*)d_in[13];
  const float* proj_w = (const float*)d_in[14];
  const float* proj_b = (const float*)d_in[15];
  const float* fc1_w = (const float*)d_in[16];
  const float* fc1_b = (const float*)d_in[17];
  const float* fc2_w = (const float*)d_in[18];
  const float* fc2_b = (const float*)d_in[19];
  const float* ad_down_w = (const float*)d_in[20];
  const float* ad_down_b = (const float*)d_in[21];
  const float* ad_up_w   = (const float*)d_in[22];
  const float* ad_up_b   = (const float*)d_in[23];
  const float* ad1_down_w = (const float*)d_in[24];
  const float* ad1_down_b = (const float*)d_in[25];
  const float* ad1_up_w   = (const float*)d_in[26];
  const float* ad1_up_b   = (const float*)d_in[27];
  const float* prompt_emb = (const float*)d_in[28];
  const float* bn_g   = (const float*)d_in[29];
  const float* bn_b   = (const float*)d_in[30];
  const float* bn_mean = (const float*)d_in[31];
  const float* bn_var  = (const float*)d_in[32];
  const float* qkv1_w  = (const float*)d_in[33];
  const float* proj1_w = (const float*)d_in[34];
  const float* proj1_b = (const float*)d_in[35];
  const float* ln3_g = (const float*)d_in[36];
  const float* ln3_b = (const float*)d_in[37];

  float* out_x = (float*)d_out;
  float* out_attn = out_x + (size_t)Bc * (G1c + 1) * Dc;

  // workspace layout (float granularity; all offsets 16B-aligned)
  float* wsf = (float*)d_ws;
  size_t off = 0;
  auto alloc  = [&](size_t nf) { float* p = wsf + off; off += nf; return p; };
  auto alloch = [&](size_t nh) { return (_Float16*)alloc(nh / 2); };  // nh even

  float*     xcat      = alloc((size_t)M1c * Dc);
  _Float16*  xn16      = alloch((size_t)M1c * Dc);        // ln1/ln2 out (reused)
  float*     qkv       = alloc((size_t)M1c * 3 * Dc);
  _Float16*  attnout16 = alloch((size_t)M1c * Dc);
  float*     x1        = alloc((size_t)M1c * Dc);
  _Float16*  h116      = alloch((size_t)M1c * 4 * Dc);
  float*     xfn       = alloc((size_t)M1c * Dc);
  _Float16*  xfn16     = alloch((size_t)M1c * Dc);
  _Float16*  adh16     = alloch((size_t)M1c * 32);        // K-padded adapter hidden
  float*     x2        = alloc((size_t)M1c * Dc);
  float*     xneigh    = alloc((size_t)M2c * Dc);
  _Float16*  xn316     = alloch((size_t)M2c * Dc);
  float*     qkv1      = alloc((size_t)M2c * 3 * Dc);
  _Float16*  attn1o16  = alloch((size_t)M2c * Dc);
  float*     xneigh2   = alloc((size_t)M2c * Dc);
  float*     xcent     = alloc((size_t)BSc * Dc);
  float*     visx      = alloc((size_t)BSc * Dc);
  float*     prop      = alloc((size_t)Bc * G1c * Dc);
  float*     x3        = alloc((size_t)M1c * Dc);
  _Float16*  x316      = alloch((size_t)M1c * Dc);
  float*     y         = alloc((size_t)M1c * Dc);
  // f16 weights
  _Float16* qkv_w16  = alloch((size_t)3 * Dc * Dc);
  _Float16* proj_w16 = alloch((size_t)Dc * Dc);
  _Float16* fc1_w16  = alloch((size_t)4 * Dc * Dc);
  _Float16* fc2_w16  = alloch((size_t)Dc * 4 * Dc);
  _Float16* addw16   = alloch((size_t)16 * Dc);
  _Float16* ad1dw16  = alloch((size_t)16 * Dc);
  _Float16* qkv1_w16 = alloch((size_t)3 * Dc * Dc);
  _Float16* proj1_w16 = alloch((size_t)Dc * Dc);
  _Float16* wup16    = alloch((size_t)Dc * 32);           // ad_up_w  padded 16->32
  _Float16* w1up16   = alloch((size_t)Dc * 32);           // ad1_up_w padded 16->32
  (void)ws_size; (void)in_sizes; (void)n_in; (void)out_size;

  const int TPB = 256;
  auto cvt = [&](const float* src, _Float16* dst, int n) {
    f32to16_kernel<<<(n + TPB - 1) / TPB, TPB, 0, stream>>>(src, dst, n);
  };

  // ---- per-call prep: weight conversion + padding ----
  cvt(qkv_w,  qkv_w16,  3 * Dc * Dc);
  cvt(proj_w, proj_w16, Dc * Dc);
  cvt(fc1_w,  fc1_w16,  4 * Dc * Dc);
  cvt(fc2_w,  fc2_w16,  Dc * 4 * Dc);
  cvt(ad_down_w,  addw16,  16 * Dc);
  cvt(ad1_down_w, ad1dw16, 16 * Dc);
  cvt(qkv1_w,  qkv1_w16,  3 * Dc * Dc);
  cvt(proj1_w, proj1_w16, Dc * Dc);
  padw_kernel<<<(Dc * 32 + TPB - 1) / TPB, TPB, 0, stream>>>(ad_up_w, wup16, Dc);
  padw_kernel<<<(Dc * 32 + TPB - 1) / TPB, TPB, 0, stream>>>(ad1_up_w, w1up16, Dc);
  zero_kernel<<<(M1c * 16 + TPB - 1) / TPB, TPB, 0, stream>>>((float*)adh16, M1c * 16);

  // ---- Stage 1: prompt insert + MHA + MLP + adapter ----
  build_xcat_kernel<<<(M1c * Dc + TPB - 1) / TPB, TPB, 0, stream>>>(x, prompt_emb, xcat);
  layernorm_kernel<<<M1c, 128, 0, stream>>>(xcat, ln1_g, ln1_b, xn16, M1c);
  launch_gemm(xn16, Dc, qkv_w16, Dc, nullptr, qkv, nullptr, 3 * Dc,
              nullptr, 0.f, nullptr, 0.f, M1c, 3 * Dc, Dc, 1.0f, 0, stream);
  attn1_kernel<<<Bc * Hc * N1c, 128, 0, stream>>>(qkv, out_attn, attnout16);
  launch_gemm(attnout16, Dc, proj_w16, Dc, proj_b, x1, nullptr, Dc,
              xcat, 1.0f, nullptr, 0.f, M1c, Dc, Dc, 1.0f, 0, stream);
  layernorm_kernel<<<M1c, 128, 0, stream>>>(x1, ln2_g, ln2_b, xn16, M1c);
  launch_gemm(xn16, Dc, fc1_w16, Dc, fc1_b, nullptr, h116, 4 * Dc,
              nullptr, 0.f, nullptr, 0.f, M1c, 4 * Dc, Dc, 1.0f, 1, stream);
  launch_gemm(h116, 4 * Dc, fc2_w16, 4 * Dc, fc2_b, xfn, xfn16, Dc,
              nullptr, 0.f, nullptr, 0.f, M1c, Dc, 4 * Dc, 1.0f, 0, stream);
  launch_gemm(xfn16, Dc, addw16, Dc, ad_down_b, nullptr, adh16, 32,
              nullptr, 0.f, nullptr, 0.f, M1c, 16, Dc, 1.0f, 1, stream);
  launch_gemm(adh16, 32, wup16, 32, ad_up_b, x2, nullptr, Dc,
              xfn, 1.0f, x1, 1.0f, M1c, Dc, 32, 0.7f, 0, stream);

  // ---- Stage 2: gather + LN + MHA over groups ----
  gather_kernel<<<(M2c * Dc + TPB - 1) / TPB, TPB, 0, stream>>>(x2, idx, xneigh, M2c);
  gather_kernel<<<(BSc * Dc + TPB - 1) / TPB, TPB, 0, stream>>>(x2, center_idx, xcent, BSc);
  layernorm_kernel<<<M2c, 128, 0, stream>>>(xneigh, ln3_g, ln3_b, xn316, M2c);
  launch_gemm(xn316, Dc, qkv1_w16, Dc, nullptr, qkv1, nullptr, 3 * Dc,
              nullptr, 0.f, nullptr, 0.f, M2c, 3 * Dc, Dc, 1.0f, 0, stream);
  attn2_kernel<<<BSc * Hc, 256, 0, stream>>>(qkv1, attn1o16);
  launch_gemm(attn1o16, Dc, proj1_w16, Dc, proj1_b, xneigh2, nullptr, Dc,
              xneigh, 1.0f, nullptr, 0.f, M2c, Dc, Dc, 1.0f, 0, stream);

  // ---- max-pool + BN + GELU + centers ----
  maxbn_kernel<<<(BSc * Dc + TPB - 1) / TPB, TPB, 0, stream>>>(
      xneigh2, bn_g, bn_b, bn_mean, bn_var, xcent, visx);

  // ---- propagate (all-32-neighbor weighted mean; no sort needed) ----
  propagate_kernel<<<Bc * G1c, 128, 0, stream>>>(center1, center2, x2, visx, prop);

  // ---- final adapter + residual + output ----
  build_x3_kernel<<<(M1c * Dc + TPB - 1) / TPB, TPB, 0, stream>>>(x2, prop, x3, x316);
  launch_gemm(x316, Dc, ad1dw16, Dc, ad1_down_b, nullptr, adh16, 32,
              nullptr, 0.f, nullptr, 0.f, M1c, 16, Dc, 1.0f, 1, stream);
  launch_gemm(adh16, 32, w1up16, 32, ad1_up_b, y, nullptr, Dc,
              x3, 1.0f, nullptr, 0.f, M1c, Dc, 32, 1.0f, 0, stream);
  final_copy_kernel<<<(Bc * (G1c + 1) * Dc + TPB - 1) / TPB, TPB, 0, stream>>>(y, out_x);
}
